// Llama4TextExperts_37589553774969
// MI455X (gfx1250) — compile-verified, matrix operand order checked by reading, not run on hardware
//
#include <hip/hip_runtime.h>
#include <cstdint>

// Problem constants (match reference)
#define E_   16
#define H_   2048
#define F_   2048      // INTERMEDIATE
#define TPE  1024      // tokens per expert (16384 / 16)

// Tiling: block tile 128(M) x 128(N), K-chunk 32; 8 waves, each 64x32 (GEMM1
// additionally computes the matching "up" 64x32, fused SwiGLU in registers).
#define BM   128
#define BN   128
#define BK   32
#define KT1  (H_ / BK)  // 64 K-steps for GEMM1
#define KT2  (F_ / BK)  // 64 K-steps for GEMM2

#define GLOBAL_AS __attribute__((address_space(1)))
#define LDS_AS    __attribute__((address_space(3)))

#if __has_builtin(__builtin_amdgcn_global_load_async_to_lds_b128) && \
    __has_builtin(__builtin_amdgcn_s_wait_asynccnt)
#define USE_ASYNC 1
#else
#define USE_ASYNC 0
#endif

typedef __attribute__((ext_vector_type(16))) _Float16 v16h;
typedef __attribute__((ext_vector_type(8)))  float    v8f;
typedef __attribute__((ext_vector_type(4)))  int      v4i;

// fp32 pair -> packed f16x2 via v_cvt_pk_rtz_f16_f32 (single VALU op).
__device__ __forceinline__ uint32_t pack_f16x2(float a, float b) {
  auto r = __builtin_amdgcn_cvt_pkrtz(a, b);
  uint32_t d;
  __builtin_memcpy(&d, &r, 4);
  return d;
}
__device__ __forceinline__ uint16_t cvt_f16(float a) {
  _Float16 h = (_Float16)a;          // v_cvt_f16_f32
  uint16_t d;
  __builtin_memcpy(&d, &h, 2);
  return d;
}

// Build a 16x32 (A, row-per-lane) or 32x16 (B, column-per-lane) f16 WMMA
// fragment from an LDS tile stored K-contiguous (stride BK halves).
// Per ISA layout: chunk0 = K[hi*8 .. +7], chunk1 = K[16+hi*8 .. +7], hi=lane>>4.
__device__ __forceinline__ v16h load_frag(const uint16_t* base, int off) {
  union { uint4 q[2]; v16h v; } f;
  f.q[0] = *(const uint4*)(base + off);
  f.q[1] = *(const uint4*)(base + off + 16);
  return f.v;
}

__device__ __forceinline__ void async_fence() {
#if USE_ASYNC
  __builtin_amdgcn_s_wait_asynccnt(0);
#endif
}

// ---------------------------------------------------------------------------
// GEMM1 + fused SwiGLU:  Hact[e,t,j] = up * silu(gate)
// ---------------------------------------------------------------------------
__global__ __launch_bounds__(256) void k_gemm1_swiglu(
    const float* __restrict__ X, const float* __restrict__ Wgu,
    uint16_t* __restrict__ Hact) {
  __shared__ __align__(16) uint16_t sA [2][BM * BK];   // [row][k]
  __shared__ __align__(16) uint16_t sBg[2][BN * BK];   // [n][k] (transposed)
  __shared__ __align__(16) uint16_t sBu[2][BN * BK];

  const int tid   = threadIdx.x;
  const int e     = blockIdx.z;
  const int mBase = blockIdx.y * BM;
  const int nBase = blockIdx.x * BN;

  const float* Xrow  = X   + (size_t)(e * TPE + mBase) * H_;
  const float* Wbase = Wgu + (size_t)e * H_ * (2 * F_) + nBase;  // +F_ => up

  const int lane = tid & 31, wave = tid >> 5;
  const int wm = wave & 1;      // 2 wave-rows of 64
  const int wn = wave >> 1;     // 4 wave-cols of 32
  const int hl = lane & 15, hi = lane >> 4;

  const v8f vzero = {0.f, 0.f, 0.f, 0.f, 0.f, 0.f, 0.f, 0.f};
  v8f accG[4][2], accU[4][2];
#pragma unroll
  for (int i = 0; i < 4; ++i)
#pragma unroll
    for (int j = 0; j < 2; ++j) { accG[i][j] = vzero; accU[i][j] = vzero; }

  auto stage = [&](int buf, int kt) {
    const int k0 = kt * BK;
    {  // A tile: 128x32 f32. Each thread: one 16-float row run -> 2 ds_b128.
      const int row = tid >> 1, ch = (tid & 1) * 16;
      const float4* src = (const float4*)(Xrow + (size_t)row * H_ + k0 + ch);
      const float4 f0 = src[0], f1 = src[1], f2 = src[2], f3 = src[3];
      const uint4 q0 = make_uint4(pack_f16x2(f0.x, f0.y), pack_f16x2(f0.z, f0.w),
                                  pack_f16x2(f1.x, f1.y), pack_f16x2(f1.z, f1.w));
      const uint4 q1 = make_uint4(pack_f16x2(f2.x, f2.y), pack_f16x2(f2.z, f2.w),
                                  pack_f16x2(f3.x, f3.y), pack_f16x2(f3.z, f3.w));
      *(uint4*)&sA[buf][row * BK + ch]     = q0;
      *(uint4*)&sA[buf][row * BK + ch + 8] = q1;
      if (kt + 1 < KT1) __builtin_prefetch((const void*)(src + 8), 0, 3);
    }
    {  // B tiles: 32x128 f32 each. Thread owns 2 columns x 8 K: 8 coalesced
       // b64 loads per matrix, pack in regs, 1 ds_b128 per column.
      const int n  = (tid & 63) * 2;
      const int ks = (tid >> 6) * 8;
      float2 g[8], u[8];
#pragma unroll
      for (int j = 0; j < 8; ++j) {
        const float* src = Wbase + (size_t)(k0 + ks + j) * (2 * F_) + n;
        g[j] = *(const float2*)(src);
        u[j] = *(const float2*)(src + F_);
      }
      const uint4 g0 = make_uint4(pack_f16x2(g[0].x, g[1].x), pack_f16x2(g[2].x, g[3].x),
                                  pack_f16x2(g[4].x, g[5].x), pack_f16x2(g[6].x, g[7].x));
      const uint4 g1 = make_uint4(pack_f16x2(g[0].y, g[1].y), pack_f16x2(g[2].y, g[3].y),
                                  pack_f16x2(g[4].y, g[5].y), pack_f16x2(g[6].y, g[7].y));
      const uint4 u0 = make_uint4(pack_f16x2(u[0].x, u[1].x), pack_f16x2(u[2].x, u[3].x),
                                  pack_f16x2(u[4].x, u[5].x), pack_f16x2(u[6].x, u[7].x));
      const uint4 u1 = make_uint4(pack_f16x2(u[0].y, u[1].y), pack_f16x2(u[2].y, u[3].y),
                                  pack_f16x2(u[4].y, u[5].y), pack_f16x2(u[6].y, u[7].y));
      *(uint4*)&sBg[buf][(n + 0) * BK + ks] = g0;
      *(uint4*)&sBg[buf][(n + 1) * BK + ks] = g1;
      *(uint4*)&sBu[buf][(n + 0) * BK + ks] = u0;
      *(uint4*)&sBu[buf][(n + 1) * BK + ks] = u1;
      if (kt + 1 < KT1)
        __builtin_prefetch((const void*)(Wbase + (size_t)(k0 + BK + ks) * (2 * F_) + n), 0, 3);
    }
  };

  stage(0, 0);
  __syncthreads();

  for (int kk = 0; kk < KT1; ++kk) {
    const int cur = kk & 1;
    if (kk + 1 < KT1) stage(cur ^ 1, kk + 1);   // overlap staging with WMMA

    v16h a[4], bg[2], bu[2];
#pragma unroll
    for (int tm = 0; tm < 4; ++tm)
      a[tm] = load_frag(sA[cur], (wm * 64 + tm * 16 + hl) * BK + hi * 8);
#pragma unroll
    for (int tn = 0; tn < 2; ++tn) {
      const int off = (wn * 32 + tn * 16 + hl) * BK + hi * 8;
      bg[tn] = load_frag(sBg[cur], off);
      bu[tn] = load_frag(sBu[cur], off);
    }
#pragma unroll
    for (int tm = 0; tm < 4; ++tm)
#pragma unroll
      for (int tn = 0; tn < 2; ++tn) {
        accG[tm][tn] = __builtin_amdgcn_wmma_f32_16x16x32_f16(
            false, a[tm], false, bg[tn], (short)0, accG[tm][tn], false, false);
        accU[tm][tn] = __builtin_amdgcn_wmma_f32_16x16x32_f16(
            false, a[tm], false, bu[tn], (short)0, accU[tm][tn], false, false);
      }
    __syncthreads();
  }

  // Fused SwiGLU epilogue: h = up * g * sigmoid(g); store f16 to workspace.
#pragma unroll
  for (int tm = 0; tm < 4; ++tm)
#pragma unroll
    for (int tn = 0; tn < 2; ++tn)
#pragma unroll
      for (int v = 0; v < 8; ++v) {
        const float g = accG[tm][tn][v];
        const float u = accU[tm][tn][v];
        const float hv = u * g * __builtin_amdgcn_rcpf(1.0f + __expf(-g));
        const int m = mBase + wm * 64 + tm * 16 + v + 8 * hi;
        const int n = nBase + wn * 32 + tn * 16 + hl;
        Hact[(size_t)(e * TPE + m) * F_ + n] = cvt_f16(hv);
      }
}

// ---------------------------------------------------------------------------
// GEMM2: Out[e] = Hact[e] (f16) @ Wd[e] (f32 -> f16)
// ---------------------------------------------------------------------------
__global__ __launch_bounds__(256) void k_gemm2(
    const uint16_t* __restrict__ Hact, const float* __restrict__ Wd,
    float* __restrict__ Out) {
  __shared__ __align__(16) uint16_t sA[2][BM * BK];
  __shared__ __align__(16) uint16_t sB[2][BN * BK];

  const int tid   = threadIdx.x;
  const int e     = blockIdx.z;
  const int mBase = blockIdx.y * BM;
  const int nBase = blockIdx.x * BN;

  const uint16_t* Hrow  = Hact + (size_t)(e * TPE + mBase) * F_;
  const float*    Wbase = Wd + (size_t)e * F_ * H_ + nBase;

  const int lane = tid & 31, wave = tid >> 5;
  const int wm = wave & 1, wn = wave >> 1;
  const int hl = lane & 15, hi = lane >> 4;

  const v8f vzero = {0.f, 0.f, 0.f, 0.f, 0.f, 0.f, 0.f, 0.f};
  v8f acc[4][2];
#pragma unroll
  for (int i = 0; i < 4; ++i)
#pragma unroll
    for (int j = 0; j < 2; ++j) acc[i][j] = vzero;

  auto stage = [&](int buf, int kt) {
    const int k0 = kt * BK;
    {  // A tile: 128x32 f16, pure copy. One b128 per thread; async when
       // available (GLOBAL_LOAD_ASYNC_TO_LDS_B128, ASYNCcnt-tracked).
      const int row = tid >> 1, ch = (tid & 1) * 16;
      const uint16_t* src = Hrow + (size_t)row * F_ + k0 + ch;
#if USE_ASYNC
      // Builtin signature: (v4i AS1*, v4i AS3*, imm offset, imm cpol).
      v4i* gsrc_g = (v4i*)src;                       // drop const / retype
      v4i* ldst_g = (v4i*)&sA[buf][row * BK + ch];   // generic
      __builtin_amdgcn_global_load_async_to_lds_b128(
          (GLOBAL_AS v4i*)gsrc_g, (LDS_AS v4i*)ldst_g, 0, 0);
#else
      const uint4 v = *(const uint4*)src;
      *(uint4*)&sA[buf][row * BK + ch] = v;
#endif
      if (kt + 1 < KT2) __builtin_prefetch((const void*)(src + BK), 0, 3);
    }
    {  // B tile: 32x128 f32, transpose-convert; 1 ds_b128 per column.
      const int n  = (tid & 63) * 2;
      const int ks = (tid >> 6) * 8;
      float2 w[8];
#pragma unroll
      for (int j = 0; j < 8; ++j)
        w[j] = *(const float2*)(Wbase + (size_t)(k0 + ks + j) * H_ + n);
      const uint4 w0 = make_uint4(pack_f16x2(w[0].x, w[1].x), pack_f16x2(w[2].x, w[3].x),
                                  pack_f16x2(w[4].x, w[5].x), pack_f16x2(w[6].x, w[7].x));
      const uint4 w1 = make_uint4(pack_f16x2(w[0].y, w[1].y), pack_f16x2(w[2].y, w[3].y),
                                  pack_f16x2(w[4].y, w[5].y), pack_f16x2(w[6].y, w[7].y));
      *(uint4*)&sB[buf][(n + 0) * BK + ks] = w0;
      *(uint4*)&sB[buf][(n + 1) * BK + ks] = w1;
      if (kt + 1 < KT2)
        __builtin_prefetch((const void*)(Wbase + (size_t)(k0 + BK + ks) * H_ + n), 0, 3);
    }
  };

  stage(0, 0);
  async_fence();
  __syncthreads();

  for (int kk = 0; kk < KT2; ++kk) {
    const int cur = kk & 1;
    if (kk + 1 < KT2) stage(cur ^ 1, kk + 1);

    v16h a[4], b[2];
#pragma unroll
    for (int tm = 0; tm < 4; ++tm)
      a[tm] = load_frag(sA[cur], (wm * 64 + tm * 16 + hl) * BK + hi * 8);
#pragma unroll
    for (int tn = 0; tn < 2; ++tn)
      b[tn] = load_frag(sB[cur], (wn * 32 + tn * 16 + hl) * BK + hi * 8);
#pragma unroll
    for (int tm = 0; tm < 4; ++tm)
#pragma unroll
      for (int tn = 0; tn < 2; ++tn)
        acc[tm][tn] = __builtin_amdgcn_wmma_f32_16x16x32_f16(
            false, a[tm], false, b[tn], (short)0, acc[tm][tn], false, false);

    async_fence();
    __syncthreads();
  }

#pragma unroll
  for (int tm = 0; tm < 4; ++tm)
#pragma unroll
    for (int tn = 0; tn < 2; ++tn)
#pragma unroll
      for (int v = 0; v < 8; ++v) {
        const int m = mBase + wm * 64 + tm * 16 + v + 8 * hi;
        const int n = nBase + wn * 32 + tn * 16 + hl;
        Out[(size_t)(e * TPE + m) * H_ + n] = acc[tm][tn][v];
      }
}

// ---------------------------------------------------------------------------
extern "C" void kernel_launch(void* const* d_in, const int* in_sizes, int n_in,
                              void* d_out, int out_size, void* d_ws, size_t ws_size,
                              hipStream_t stream) {
  (void)in_sizes; (void)n_in; (void)out_size; (void)ws_size;
  const float* X   = (const float*)d_in[0];   // (16384, 2048) f32
  const float* Wgu = (const float*)d_in[1];   // (16, 2048, 4096) f32
  const float* Wd  = (const float*)d_in[2];   // (16, 2048, 2048) f32
  float*    Out    = (float*)d_out;           // (16384, 2048) f32
  uint16_t* Hact   = (uint16_t*)d_ws;         // (16, 1024, 2048) f16 scratch

  dim3 blk(256);
  dim3 g1(F_ / BN, TPE / BM, E_);             // 16 x 8 x 16
  dim3 g2(H_ / BN, TPE / BM, E_);             // 16 x 8 x 16
  hipLaunchKernelGGL(k_gemm1_swiglu, g1, blk, 0, stream, X, Wgu, Hact);
  hipLaunchKernelGGL(k_gemm2,       g2, blk, 0, stream, Hact, Wd, Out);
}
